// SCM_39676907888322
// MI455X (gfx1250) — matrix operand order repaired and checked by reading, not compiled, not run on hardware
//
#include <hip/hip_runtime.h>

// ---- CDNA5 WMMA types ----
typedef __attribute__((ext_vector_type(16))) __bf16 v16bf;
typedef __attribute__((ext_vector_type(8)))  __bf16 v8bf;
typedef __attribute__((ext_vector_type(8)))  float  v8f;

#define WMMA_BF16(a, b, c) \
  __builtin_amdgcn_wmma_f32_16x16x32_bf16(false, (a), false, (b), (short)0, (c), false, false)

// problem constants
constexpr int B_  = 8;
constexpr int C_  = 256;
constexpr int CI_ = 128;
constexpr int HW_ = 2304;                       // 48*48
constexpr int PROJ_ELEMS = B_ * CI_ * HW_;      // 2,359,296 bf16 per projection

// ---------------------------------------------------------------------------
// Stage 1: six 1x1 convs as GEMMs  out[b,o,n] = sum_c W[o,c] x[b,c,n] + bias[o]
//   p = 0..5 -> {v1,k1,q1,v2,k2,q2}; v stored [b][c][n], k/q stored [b][n][c]
// One wave computes the FULL 128(o) x 16(n) column block.  Each k-step first
// builds the shared B fragment + all 8 A fragments (distinct registers), then
// issues 8 back-to-back WMMAs: no WMMA->VALU WAR hazard NOPs inside the burst.
// ---------------------------------------------------------------------------
struct QkvArgs {
  const float* x[2];
  const float* w[6];
  const float* bias[6];
  __bf16*      out[6];
};

__global__ __launch_bounds__(32)
void SCM_qkv_proj_kernel(QkvArgs args) {
  const int lane = threadIdx.x & 31;
  const int n0 = blockIdx.x * 16;
  const int p  = blockIdx.y >> 3;       // projection 0..5
  const int b  = blockIdx.y & 7;        // batch

  const float* x   = args.x[p < 3 ? 0 : 1];
  const float* wm  = args.w[p];
  const float* bi  = args.bias[p];
  __bf16*      out = args.out[p];
  const bool transposed = (p % 3) != 0; // k,q transposed; v channel-major

  const int l16   = lane & 15;
  const int hi    = lane >> 4;          // 0 or 1 (lane half)
  const int khalf = hi * 8;             // A-frag K offset pattern
  const int kb16  = hi * 16;            // unused here; B-frag also uses khalf

  const int n_b = n0 + l16;                                   // B-matrix col (n)
  const float* wrow0 = wm + (size_t)l16 * C_;                 // A row base (o-tile 0)
  const float* xcol  = x + (size_t)b * C_ * HW_ + n_b;

  v8f acc[8] = {};                      // o-tiles t: o-block = 16*t
  for (int c0 = 0; c0 < C_; c0 += 32) {
    // shared B fragment: X column, strided over c (coalesced across lanes)
    v16bf bm;
#pragma unroll
    for (int j = 0; j < 16; ++j)
      bm[j] = (__bf16)xcol[(size_t)(c0 + khalf + (j & 7) + ((j >> 3) << 4)) * HW_];
    // build all 8 A fragments (W rows) into distinct registers
    v16bf a[8];
#pragma unroll
    for (int t = 0; t < 8; ++t) {
      const float* wrow = wrow0 + (size_t)(16 * t) * C_ + c0;
#pragma unroll
      for (int j = 0; j < 8; ++j) {
        a[t][j]     = (__bf16)wrow[khalf + j];
        a[t][8 + j] = (__bf16)wrow[16 + khalf + j];
      }
    }
    // WMMA burst: 8 independent matmuls, no source rewrites in between
#pragma unroll
    for (int t = 0; t < 8; ++t)
      acc[t] = WMMA_BF16(a[t], bm, acc[t]);
  }

  const int rbase = hi * 8;             // C/D frag: rows r + 8*hi
#pragma unroll
  for (int t = 0; t < 8; ++t) {
    const int ob = 16 * t;
    if (transposed) {
      // [b][n][128]: 8 consecutive channels per lane -> one 16B store
      v8bf pack;
#pragma unroll
      for (int r = 0; r < 8; ++r)
        pack[r] = (__bf16)(acc[t][r] + bi[ob + rbase + r]);
      *(v8bf*)(out + ((size_t)(b * HW_ + n_b)) * CI_ + ob + rbase) = pack;
    } else {
      // [b][c][n]: lane-contiguous in n, strided over channel
#pragma unroll
      for (int r = 0; r < 8; ++r) {
        const int o = ob + rbase + r;
        out[((size_t)(b * CI_ + o)) * HW_ + n_b] = (__bf16)(acc[t][r] + bi[o]);
      }
    }
  }
}

// ---------------------------------------------------------------------------
// Stage 2: fused sigmoid-gated cross attention.
//   Fu[b,c,n] = sum_m sigmoid( sum_k Q[b,k,n] K[b,k,m] ) * V[b,c,m]
// Workgroup = 8 waves, one 32-wide n-tile, all 128 channels.
// Q fragments (both 16-n halves, full C=128) hoisted into registers.
// Per 128-m strip: phase A preloads 4 K fragments then bursts 8 S-WMMAs;
// sigmoid(S) is scattered to LDS pre-shuffled into WMMA B-fragment layout;
// phase B preloads 4 V fragments then bursts 8 Fu-WMMAs (LDS B-frags inline).
// ---------------------------------------------------------------------------
struct AttnArgs {
  const __bf16* Q[2];
  const __bf16* K[2];
  const __bf16* V[2];
  float*        y[2];   // y1 / y2 base, layout (B, 384, 2304)
};

__global__ __launch_bounds__(256)
void SCM_attn_fuse_kernel(AttnArgs args) {
  // [n-half][m-chunk][lane][16 bf16] = 8 KB, rows are 32B for v16bf loads
  __shared__ alignas(32) __bf16 Sld[2][4][32][16];

  const int lane = threadIdx.x & 31;
  const int w    = threadIdx.x >> 5;    // wave 0..7
  const int n0   = blockIdx.x * 32;
  const int b    = blockIdx.y;
  const int pr   = blockIdx.z;          // 0: Fu1(Q1,K2,V2)  1: Fu2(Q2,K1,V1)

  const __bf16* Qt = args.Q[pr];
  const __bf16* Kt = args.K[pr];
  const __bf16* V  = args.V[pr];
  float*        y  = args.y[pr];

  const int l16   = lane & 15;
  const int hi    = lane >> 4;
  const int khalf = hi * 8;
  const int kb16  = hi * 16;

  // ---- hoist Q fragments: invariant across the whole m loop ----
  v16bf qf[2][4];                       // [n-half][ck/32]
#pragma unroll
  for (int h = 0; h < 2; ++h) {
    const __bf16* qrow = Qt + ((size_t)(b * HW_) + (n0 + 16 * h + l16)) * CI_;
#pragma unroll
    for (int c4 = 0; c4 < 4; ++c4) {
      const v8bf qlo = *(const v8bf*)(qrow + 32 * c4 + kb16);
      const v8bf qhi = *(const v8bf*)(qrow + 32 * c4 + kb16 + 8);
#pragma unroll
      for (int j = 0; j < 8; ++j) { qf[h][c4][j] = qlo[j]; qf[h][c4][8 + j] = qhi[j]; }
    }
  }

  // phase-B A-frag rows: this wave's channel block
  const __bf16* vrow = V + ((size_t)(b * CI_) + (w * 16 + l16)) * HW_;

  // LDS scatter targets for this wave's S tiles (B-fragment layout):
  // S element (k_in_32chunk, n_local) lives at Sld[h][mk][n + 16*(k>=16)][k%16]
  const int mk_w  = w >> 1;             // which 32-m chunk this wave feeds
  const int lhi_w = (w & 1) * 16;       // low/high half of the chunk's K range
  const int t_w   = l16;                // our fixed k%16 (= D-frag column)

  v8f facc[2] = {};

  for (int mc = 0; mc < HW_; mc += 128) {
    // ------------- Phase A: two S tiles, m0 = mc + 16*w -------------
    const int m0 = mc + 16 * w;
    const __bf16* krow = Kt + ((size_t)(b * HW_) + (m0 + l16)) * CI_;
    // preload all 4 K fragments, then burst 8 WMMAs
    v16bf kf[4];
#pragma unroll
    for (int c4 = 0; c4 < 4; ++c4)
      kf[c4] = *(const v16bf*)(krow + 32 * c4 + kb16);  // 32B contiguous
    v8f sacc[2] = {};
#pragma unroll
    for (int c4 = 0; c4 < 4; ++c4) {
      sacc[0] = WMMA_BF16(qf[0][c4], kf[c4], sacc[0]);
      sacc[1] = WMMA_BF16(qf[1][c4], kf[c4], sacc[1]);
    }
    // sigmoid in f32, scatter bf16 into B-fragment layout in LDS
#pragma unroll
    for (int h = 0; h < 2; ++h)
#pragma unroll
      for (int r = 0; r < 8; ++r) {
        const float sv = 1.0f / (1.0f + __expf(-sacc[h][r]));
        Sld[h][mk_w][(r + hi * 8) + lhi_w][t_w] = (__bf16)sv;
      }
    __syncthreads();

    // ------------- Phase B: Fu += V * S^T (contract 128 m) -------------
    // preload all 4 V fragments, then burst 8 WMMAs (LDS B-frags inline)
    v16bf vf[4];
#pragma unroll
    for (int mk = 0; mk < 4; ++mk) {
      const int mbase = mc + mk * 32;
      const v8bf vlo = *(const v8bf*)(vrow + mbase + kb16);
      const v8bf vhi = *(const v8bf*)(vrow + mbase + kb16 + 8);
#pragma unroll
      for (int j = 0; j < 8; ++j) { vf[mk][j] = vlo[j]; vf[mk][8 + j] = vhi[j]; }
    }
#pragma unroll
    for (int mk = 0; mk < 4; ++mk) {
      facc[0] = WMMA_BF16(vf[mk], *(const v16bf*)(&Sld[0][mk][lane][0]), facc[0]);
      facc[1] = WMMA_BF16(vf[mk], *(const v16bf*)(&Sld[1][mk][lane][0]), facc[1]);
    }
    __syncthreads();

    if (mc + 128 < HW_) {  // warm next strip (global_prefetch_b8)
      __builtin_prefetch(Kt + ((size_t)(b * HW_) + (mc + 128 + 16 * w + l16)) * CI_, 0, 1);
      __builtin_prefetch(vrow + mc + 128, 0, 1);
    }
  }

  // store Fu tiles into concat channels [256..384) of y
#pragma unroll
  for (int h = 0; h < 2; ++h)
#pragma unroll
    for (int r = 0; r < 8; ++r) {
      const int ch = 256 + w * 16 + hi * 8 + r;
      y[((size_t)(b * 384) + ch) * HW_ + n0 + 16 * h + l16] = facc[h][r];
    }
}

// ---------------------------------------------------------------------------
// Stage 3: concat copy x1 -> y1[:, :256], x2 -> y2[:, :256] (float4)
// ---------------------------------------------------------------------------
__global__ __launch_bounds__(256)
void SCM_copy_x_kernel(const float4* __restrict__ x1, const float4* __restrict__ x2,
                       float4* __restrict__ y) {
  const size_t XB  = (size_t)C_ * HW_ / 4;      // 147456 vec4 per batch of x
  const size_t YB  = (size_t)384 * HW_ / 4;     // 221184 vec4 per batch of y
  const size_t PER = (size_t)B_ * XB;
  size_t tid = (size_t)blockIdx.x * blockDim.x + threadIdx.x;
  if (tid >= 2 * PER) return;
  const int s    = tid >= PER;
  const size_t r = tid - (size_t)s * PER;
  const size_t b = r / XB;
  const size_t o = r % XB;
  y[(size_t)s * (B_ * YB) + b * YB + o] = (s ? x2 : x1)[r];
}

// ---------------------------------------------------------------------------
extern "C" void kernel_launch(void* const* d_in, const int* in_sizes, int n_in,
                              void* d_out, int out_size, void* d_ws, size_t ws_size,
                              hipStream_t stream) {
  const float* x1 = (const float*)d_in[0];
  const float* x2 = (const float*)d_in[1];
  __bf16* ws = (__bf16*)d_ws;   // 6 * PROJ_ELEMS bf16 = 28.3 MB

  QkvArgs qa;
  qa.x[0] = x1;
  qa.x[1] = x2;
  for (int p = 0; p < 6; ++p) {     // p: 0=v1,1=k1,2=q1,3=v2,4=k2,5=q2
    qa.w[p]    = (const float*)d_in[2 + 2 * p];
    qa.bias[p] = (const float*)d_in[3 + 2 * p];
    qa.out[p]  = ws + (size_t)p * PROJ_ELEMS;
  }
  dim3 g1(HW_ / 16, 6 * B_);
  SCM_qkv_proj_kernel<<<g1, 32, 0, stream>>>(qa);

  AttnArgs aa;
  aa.Q[0] = ws + 2 * (size_t)PROJ_ELEMS;  // q1
  aa.K[0] = ws + 4 * (size_t)PROJ_ELEMS;  // k2
  aa.V[0] = ws + 3 * (size_t)PROJ_ELEMS;  // v2
  aa.Q[1] = ws + 5 * (size_t)PROJ_ELEMS;  // q2
  aa.K[1] = ws + 1 * (size_t)PROJ_ELEMS;  // k1
  aa.V[1] = ws + 0 * (size_t)PROJ_ELEMS;  // v1
  float* y = (float*)d_out;
  aa.y[0] = y;
  aa.y[1] = y + (size_t)B_ * 384 * HW_;
  dim3 g2(HW_ / 32, B_, 2);
  SCM_attn_fuse_kernel<<<g2, 256, 0, stream>>>(aa);

  const size_t total4 = 2ull * B_ * C_ * HW_ / 4;
  SCM_copy_x_kernel<<<(unsigned)((total4 + 255) / 256), 256, 0, stream>>>(
      (const float4*)x1, (const float4*)x2, (float4*)d_out);
}
